// DisentangleStaticNoiseLoss_47425028883083
// MI455X (gfx1250) — matrix operand order repaired and checked by reading, not compiled, not run on hardware
//
#include <hip/hip_runtime.h>
#include <hip/hip_bf16.h>

#define N_T 8
#define BS  1024
#define DD  128
#define KK  (N_T * BS)        // 8192
#define JT_TOTAL (KK / 16)    // 512 column tiles
#define TEMP_INV 2.0f         // 1 / 0.5
#define SHIFT    2.0f         // logits are 2*cos in [-2,2]
#define EPSN     1e-8f
#define TILE_ELEMS (16 * DD)  // 2048 bf16 = 4096 B per (buf, h/l) region

typedef __attribute__((ext_vector_type(16))) __bf16 v16bf;
typedef __attribute__((ext_vector_type(8)))  __bf16 bf16x8;
typedef __attribute__((ext_vector_type(4)))  __bf16 bf16x4;
typedef __attribute__((ext_vector_type(8)))  float  v8f;
typedef __attribute__((ext_vector_type(4)))  unsigned int u32x4;
typedef __attribute__((ext_vector_type(8)))  int i32x8;
typedef __attribute__((ext_vector_type(4)))  int i32x4;

__device__ inline __bf16 f2bf(float f) {
    unsigned u = __float_as_uint(f);
    unsigned r = (u + 0x7FFFu + ((u >> 16) & 1u)) >> 16;   // RNE
    unsigned short s = (unsigned short)r;
    return __builtin_bit_cast(__bf16, s);
}
__device__ inline float bf2f(__bf16 h) {
    unsigned short s = __builtin_bit_cast(unsigned short, h);
    return __uint_as_float(((unsigned)s) << 16);
}

// Escape the LDS pointer into a memory-clobber asm: the capture makes the
// asm a potential writer of the buffer, so the TDM's (invisible) LDS writes
// can't be optimized into undef loads, and loads order after the barrier.
__device__ inline void lds_publish(const void* p) {
    asm volatile("" : : "v"(p) : "memory");
}

// ---------------------------------------------------------------------------
// Kernel 1: row-normalize x (8192 x 128), emit z (f32) and split-bf16 z=h+l
// ---------------------------------------------------------------------------
__global__ void k_normalize(const float* __restrict__ x,
                            float* __restrict__ z,
                            __bf16* __restrict__ zh,
                            __bf16* __restrict__ zl) {
    int wave = (int)((blockIdx.x * blockDim.x + threadIdx.x) >> 5);
    int lane = (int)(threadIdx.x & 31u);
    if (wave >= KK) return;
    float4 v = ((const float4*)(x + (size_t)wave * DD))[lane];
    float ss = v.x * v.x + v.y * v.y + v.z * v.z + v.w * v.w;
#pragma unroll
    for (int m = 16; m >= 1; m >>= 1) ss += __shfl_xor(ss, m, 32);
    float inv = 1.0f / fmaxf(sqrtf(ss), EPSN);
    v.x *= inv; v.y *= inv; v.z *= inv; v.w *= inv;
    ((float4*)(z + (size_t)wave * DD))[lane] = v;
    float t[4] = {v.x, v.y, v.z, v.w};
    bf16x4 h, l;
#pragma unroll
    for (int i = 0; i < 4; ++i) {
        __bf16 hh = f2bf(t[i]);
        h[i] = hh;
        l[i] = f2bf(t[i] - bf2f(hh));
    }
    ((bf16x4*)(zh + (size_t)wave * DD))[lane] = h;
    ((bf16x4*)(zl + (size_t)wave * DD))[lane] = l;
}

// ---------------------------------------------------------------------------
// WMMA operand layouts as loaded by this code (row-major z in memory).
// A tile (16x32 bf16): lane holds row = base + lane%16,
//   K = sel*16..+16 (contiguous, matching B)   (sel = lane/16)
// B tile (32x16 bf16): lane holds col = base + lane%16, K = sel*16..+16
// ---------------------------------------------------------------------------
__device__ inline v16bf load_opA(const __bf16* __restrict__ rowbase, int k0, int sel) {
    bf16x8 lo = *(const bf16x8*)(rowbase + k0 + sel * 16);
    bf16x8 hi = *(const bf16x8*)(rowbase + k0 + sel * 16 + 8);
    v16bf r;
#pragma unroll
    for (int i = 0; i < 8; ++i) { r[i] = lo[i]; r[i + 8] = hi[i]; }
    return r;
}
__device__ inline v16bf load_opB_lds(const __bf16* p) {   // p -> lane's row in LDS
    bf16x8 lo = *(const bf16x8*)(p);
    bf16x8 hi = *(const bf16x8*)(p + 8);
    v16bf r;
#pragma unroll
    for (int i = 0; i < 8; ++i) { r[i] = lo[i]; r[i + 8] = hi[i]; }
    return r;
}

// ---------------------------------------------------------------------------
// TDM: stage one 16-row x 256B tile (of an 8192x256B tensor) into LDS.
// D# per ISA ch.8: 2D tensor, 8-byte elements; group0 = {count/type, lds,
// global_addr}; group1 = dims/tiles/strides. 6-arg builtin on this toolchain.
// ---------------------------------------------------------------------------
__device__ inline void tdm_stage(const __bf16* gbase, int row0, unsigned lds_off) {
    unsigned long long ga =
        (unsigned long long)(__UINTPTR_TYPE__)(gbase + (size_t)row0 * DD);
    u32x4 g0;
    g0[0] = 1u;                                   // count=1, user mode, no gather
    g0[1] = lds_off;                              // lds_addr (bytes)
    g0[2] = (unsigned)(ga & 0xFFFFFFFFu);         // global_addr[31:0]
    g0[3] = (unsigned)((ga >> 32) & 0x01FFFFFFu)  // global_addr[56:32]
          | (2u << 30);                           // type=2 ("image")
    i32x8 g1;
    g1[0] = (3 << 16);                            // wg_mask=0, data_size=3 (8B)
    g1[1] = (int)((32u & 0xFFFFu) << 16);         // tensor_dim0=32 [63:48]
    g1[2] = (int)((8192u & 0xFFFFu) << 16);       // dim0 hi=0 | tensor_dim1 lo
    g1[3] = (int)((8192u >> 16) | (32u << 16));   // dim1 hi | tile_dim0=32
    g1[4] = 16;                                   // tile_dim1=16, tile_dim2=0
    g1[5] = 32;                                   // tensor_dim0_stride=32 units
    g1[6] = 0;
    g1[7] = 0;
    i32x4 zg4 = {0, 0, 0, 0};
    i32x8 zg8 = {0, 0, 0, 0, 0, 0, 0, 0};
    __builtin_amdgcn_tensor_load_to_lds(g0, g1, zg4, zg4, zg8, 0);
}

// ---------------------------------------------------------------------------
// Kernel 2: fused Gram + negatives sum-exp.
// 4 waves/block, one 16-row i-tile per wave. B tiles staged into DYNAMIC LDS
// by the Tensor Data Mover (double-buffered, overlapped with WMMA), consumed
// by all waves via ds_load. Split-bf16 GEMM: S ~= h*h^T + h*l^T + l*h^T.
// Fixed-shift softmax: Sneg[i] = sum_{j%BS != i%BS} exp(2*cos - 2)
// ---------------------------------------------------------------------------
__global__ void __launch_bounds__(128)
k_gram_softmax(const __bf16* __restrict__ zh,
               const __bf16* __restrict__ zl,
               float* __restrict__ Sneg) {
    // dynamic LDS: [dbuf][h/l][16 rows * 128] = 16 KB, starts at LDS offset 0
    extern __shared__ __bf16 lbuf[];

    int lane = (int)(threadIdx.x & 31u);
    int wid  = (int)(threadIdx.x >> 5);           // 0..3
    int wave = (int)blockIdx.x * 4 + wid;
    int i0   = wave * 16;                         // 512 i-tiles total
    int sel  = lane >> 4;
    int arow = i0 + (lane & 15);
    bool issuer = (wid == 0);

    const __bf16* ahb = zh + (size_t)arow * DD;
    const __bf16* alb = zl + (size_t)arow * DD;
    v16bf aH[4], aL[4];
#pragma unroll
    for (int c = 0; c < 4; ++c) {
        aH[c] = load_opA(ahb, 32 * c, sel);
        aL[c] = load_opA(alb, 32 * c, sel);
    }

    float E[8];
#pragma unroll
    for (int v = 0; v < 8; ++v) E[v] = 0.0f;
    int ms = (lane & 15) - 8 * sel;               // masked C slot (if 0..7)

    // prologue: stage tile 0 into buffer 0
    if (issuer) {
        tdm_stage(zh, 0, 0u);
        tdm_stage(zl, 0, 4096u);
        __builtin_amdgcn_s_wait_tensorcnt(0);
    }
    __syncthreads();
    lds_publish(lbuf);                            // TDM wrote lbuf: force reloads

    int cur = 0;
    for (int jt = 0; jt < JT_TOTAL; ++jt) {
        if (issuer && (jt + 1 < JT_TOTAL)) {      // async-stage next tile
            unsigned nb = (unsigned)(cur ^ 1) * 8192u;
            tdm_stage(zh, (jt + 1) * 16, nb);
            tdm_stage(zl, (jt + 1) * 16, nb + 4096u);
        }

        const __bf16* bh = lbuf + (cur * 2 + 0) * TILE_ELEMS + (lane & 15) * DD;
        const __bf16* bl = lbuf + (cur * 2 + 1) * TILE_ELEMS + (lane & 15) * DD;
        v8f c = {};
#pragma unroll
        for (int ck = 0; ck < 4; ++ck) {
            v16bf bH = load_opB_lds(bh + 32 * ck + sel * 16);
            v16bf bL = load_opB_lds(bl + 32 * ck + sel * 16);
            c = __builtin_amdgcn_wmma_f32_16x16x32_bf16(false, aH[ck], false, bH,
                                                        (short)0, c, false, false);
            c = __builtin_amdgcn_wmma_f32_16x16x32_bf16(false, aH[ck], false, bL,
                                                        (short)0, c, false, false);
            c = __builtin_amdgcn_wmma_f32_16x16x32_bf16(false, aL[ck], false, bH,
                                                        (short)0, c, false, false);
        }

        // excluded columns (self + positives) = tile diagonal of BS-aligned tiles
        int j0 = jt * 16;
        bool aligned = (((unsigned)(i0 - j0) & (BS - 1)) == 0u);
#pragma unroll
        for (int v = 0; v < 8; ++v) {
            float s = c[v] * TEMP_INV;            // logit = 2*cos
            float e = __expf(s - SHIFT);          // <= 1, stable fixed shift
            if (aligned && v == ms) e = 0.0f;
            E[v] += e;
        }

        if (issuer) __builtin_amdgcn_s_wait_tensorcnt(0);
        __syncthreads();
        lds_publish(lbuf);                        // next buffer now valid
        cur ^= 1;
    }

    // reduce across the 16 lanes sharing each output row (xor stays in-half)
#pragma unroll
    for (int v = 0; v < 8; ++v) {
        float e = E[v];
#pragma unroll
        for (int m = 1; m <= 8; m <<= 1) e += __shfl_xor(e, m, 32);
        E[v] = e;
    }
    if ((lane & 15) == 0) {
#pragma unroll
        for (int v = 0; v < 8; ++v)
            Sneg[i0 + 8 * sel + v] = E[v];        // row = i0 + v + 8*sel
    }
}

// ---------------------------------------------------------------------------
// Kernel 3: 7 positives per row (exact fp32 dots) + per-block loss partials.
// ---------------------------------------------------------------------------
__global__ void k_loss(const float* __restrict__ z,
                       const float* __restrict__ Sneg,
                       float* __restrict__ partial) {
    int lane = (int)(threadIdx.x & 31u);
    int i    = (int)((blockIdx.x * blockDim.x + threadIdx.x) >> 5);
    int r    = i & (BS - 1);
    int b    = i >> 10;
    float4 a = ((const float4*)(z + (size_t)i * DD))[lane];
    float sn = Sneg[i];
    float acc = 0.0f;
#pragma unroll
    for (int bp = 0; bp < N_T; ++bp) {
        if (bp == b) continue;
        int j = r + bp * BS;
        float4 bv = ((const float4*)(z + (size_t)j * DD))[lane];
        float d = a.x * bv.x + a.y * bv.y + a.z * bv.z + a.w * bv.w;
#pragma unroll
        for (int m = 16; m >= 1; m >>= 1) d += __shfl_xor(d, m, 32);
        float pos = d * TEMP_INV;
        float lse = SHIFT + logf(expf(pos - SHIFT) + sn);
        acc += (lse - pos);
    }
    __shared__ float sbuf[8];
    if (lane == 0) sbuf[threadIdx.x >> 5] = acc;
    __syncthreads();
    if (threadIdx.x == 0) {
        float s = 0.0f;
#pragma unroll
        for (int w = 0; w < 8; ++w) s += sbuf[w];
        partial[blockIdx.x] = s;
    }
}

// ---------------------------------------------------------------------------
// Kernel 4: final reduction of 1024 partials -> mean loss
// ---------------------------------------------------------------------------
__global__ void k_reduce(const float* __restrict__ partial, float* __restrict__ out) {
    __shared__ float sm[256];
    float s = 0.0f;
    for (int i = (int)threadIdx.x; i < 1024; i += 256) s += partial[i];
    sm[threadIdx.x] = s;
    __syncthreads();
    for (int stride = 128; stride > 0; stride >>= 1) {
        if ((int)threadIdx.x < stride) sm[threadIdx.x] += sm[threadIdx.x + stride];
        __syncthreads();
    }
    if (threadIdx.x == 0)
        out[0] = sm[0] * (1.0f / ((float)KK * (float)(N_T - 1)));
}

extern "C" void kernel_launch(void* const* d_in, const int* in_sizes, int n_in,
                              void* d_out, int out_size, void* d_ws, size_t ws_size,
                              hipStream_t stream) {
    (void)in_sizes; (void)n_in; (void)out_size; (void)ws_size;
    const float* x = (const float*)d_in[0];

    char* ws = (char*)d_ws;
    float*  z    = (float*) (ws);                       // 8192*128*4 = 4 MB
    __bf16* zh   = (__bf16*)(ws + 4u * 1024u * 1024u);  // 2 MB
    __bf16* zl   = (__bf16*)(ws + 6u * 1024u * 1024u);  // 2 MB
    float*  sneg = (float*) (ws + 8u * 1024u * 1024u);  // 32 KB
    float*  part = (float*) (ws + 8u * 1024u * 1024u + 32u * 1024u); // 4 KB

    // 1) normalize + split-bf16 (8192 waves)
    k_normalize<<<1024, 256, 0, stream>>>(x, z, zh, zl);
    // 2) fused Gram + neg sum-exp (512 waves, 4/block, TDM-staged B, 16KB dyn LDS)
    k_gram_softmax<<<128, 128, 16384, stream>>>(zh, zl, sneg);
    // 3) positives + per-block partials (8192 waves)
    k_loss<<<1024, 256, 0, stream>>>(z, sneg, part);
    // 4) final scalar
    k_reduce<<<1, 256, 0, stream>>>(part, (float*)d_out);
}
